// DSSMs_40905268527434
// MI455X (gfx1250) — matrix operand order repaired and checked by reading, not compiled, not run on hardware
//
#include <hip/hip_runtime.h>
#include <hip/hip_bf16.h>
#include <math.h>

typedef __attribute__((ext_vector_type(16))) _Float16 v16h;
typedef __attribute__((ext_vector_type(8)))  float    v8f;
typedef __attribute__((ext_vector_type(2)))  float    v2f;

#define NMC_PAD 112  // 100 ensemble members padded to 7 tiles of 16 (pad rows stay zero)

// ---------------- LDS strides (element units, padded to dodge 64-bank conflicts) -------------
#define XS  132   // X ensemble, f32 [112][132]
#define AS  136   // a16 staging, f16 [112][136]
#define H1S 136   // h1, f16
#define H2S 72    // h2, f16
#define H3S 40    // h3, f16
#define SS  65    // S / Cholesky factor, f32 [64][65]
#define KS  132   // Kalman gain / HP, f32 [64][132]
#define INS 68    // innovation, f32 [112][68]

// ---------------- LDS offsets (bytes); regions are time-multiplexed within one t-step --------
#define X_OFF     0
#define A_OFF     (NMC_PAD*XS*4)                  // 59136  (a16 during MLP; innov later)
#define H1_OFF    (A_OFF + NMC_PAD*AS*2)          // 89600  (h1 during MLP; S later)
#define H2_OFF    (H1_OFF + NMC_PAD*H1S*2)        // 120064 (h2+h3 during MLP; Kgain later)
#define H3_OFF    (H2_OFF + NMC_PAD*H2S*2)        // 136192
#define SMALL_OFF (H2_OFF + 64*KS*4)              // 153856
#define SM_XM    (SMALL_OFF + 0)
#define SM_QM    (SM_XM + 512)
#define SM_QV    (SM_QM + 512)
#define SM_YV    (SM_QV + 512)
#define SM_YP    (SM_YV + 256)
#define SM_H     (SM_YP + 256)
#define SM_C     (SM_H + 256)
#define SM_XL    (SM_C + 256)
#define SM_G     (SM_XL + 256)
#define SM_RED   (SM_G + 1024)
#define SM_Z     (SM_RED + 1024)
#define SM_B1    (SM_Z + 256)
#define SM_B2    (SM_B1 + 512)
#define SM_B3    (SM_B2 + 256)
#define SM_B4    (SM_B3 + 128)
#define SM_BIH   (SM_B4 + 512)
// persistent LDS copies of the fragment-swizzled f16 weights (staged once, reused 50x)
#define WL1_OFF  (SM_BIH + 1024)                  // 161408, 32768 B
#define WL2_OFF  (WL1_OFF + 32768)                // 194176, 16384 B
#define WL3_OFF  (WL2_OFF + 16384)                // 210560,  4096 B
#define WL4_OFF  (WL3_OFF + 4096)                 // 214656,  8192 B
// per-step eps tile, filled by async global->LDS loads overlapping the MLP
#define EPS_OFF  (WL4_OFF + 8192)                 // 222848, 100*128*4 = 51200 B
#define SMEM_TOTAL (EPS_OFF + 51200)              // 274048 B < 320 KB WGP LDS

// ---------------- workspace offsets (bytes) ----------------
#define WS_TAPER 0          // f32 [128][128]
#define WS_B1    65536      // f16 swizzled fragments, layer1 (32768 B)
#define WS_B2    98304      // layer2 (16384 B)
#define WS_B3    114688     // layer3 (4096 B)
#define WS_B4    118784     // layer4 (8192 B)   [B1..B4 contiguous: 61440 B]
#define WS_KL    126976     // f32 [128]
#define WS_LL    127488     // f32 [50*128]

__device__ __forceinline__ float sigmoidf_(float x) { return 1.0f / (1.0f + __expf(-x)); }

// ----- WMMA fragment loaders, f32 per CDNA5 ISA 7.12.2 layouts, f16 with contiguous k = 16*(lane/16)+s (wave32) -----
// f16 A 16x32 (MxK)
__device__ __forceinline__ v16h load_a16f(const _Float16* A, int lda, int mb, int kb, int lane) {
  int r = lane & 15, hi = lane >> 4;
  const _Float16* row = A + (mb + r) * lda + kb + hi * 16;
  v16h a;
#pragma unroll
  for (int h = 0; h < 16; ++h) a[h] = row[h];
  return a;
}
// f16 B 32x16 (KxN) from fragment-swizzled LDS copy: one contiguous 32B read per lane
__device__ __forceinline__ v16h load_b16sw(const _Float16* Bsw, int NT, int kc, int nt, int lane) {
  return *(const v16h*)(Bsw + ((kc * NT + nt) * 32 + lane) * 16);
}
// f32 A 16x4
__device__ __forceinline__ v2f load_a32f(const float* A, int lda, int mb, int kb, int lane) {
  int r = lane & 15, hi = lane >> 4;
  const float* p = A + (mb + r) * lda + kb + 2 * hi;
  v2f a; a[0] = p[0]; a[1] = p[1]; return a;
}
// f32 A 16x4 where A[o][k] = Xc[k][o] (transposed read from LDS)
__device__ __forceinline__ v2f load_a32t(const float* Xc, int ldx, int ob, int kb, int lane) {
  int r = lane & 15, hi = lane >> 4;
  const float* p = Xc + (kb + 2 * hi) * ldx + ob + r;
  v2f a; a[0] = p[0]; a[1] = p[ldx]; return a;
}
// f32 B 4x16
__device__ __forceinline__ v2f load_b32f(const float* B, int ldb, int kb, int nb, int lane) {
  int c = lane & 15, hi = lane >> 4;
  const float* p = B + (kb + 2 * hi) * ldb + nb + c;
  v2f b; b[0] = p[0]; b[1] = p[ldb]; return b;
}

// fragment-major swizzle index for B[k][n] (= W^T), NT = N/16 tiles
__device__ __forceinline__ int swz_idx(int k, int n, int NT) {
  int kc = k >> 5, koff = k & 31;
  int hi = koff >> 4, h = koff & 15;
  int nt = n >> 4, c = n & 15;
  return ((kc * NT + nt) * 32 + (c + 16 * hi)) * 16 + h;
}

// =====================  prep: Gaspari-Cohn taper + f16 swizzled weights  =====================
__global__ void dssm_prep_kernel(const float* __restrict__ W1, const float* __restrict__ W2,
                                 const float* __restrict__ W3, const float* __restrict__ W4,
                                 float* __restrict__ taper,
                                 _Float16* __restrict__ B1, _Float16* __restrict__ B2,
                                 _Float16* __restrict__ B3, _Float16* __restrict__ B4) {
  int idx = blockIdx.x * blockDim.x + threadIdx.x;
  if (idx < 128 * 128) {
    int i = idx >> 7, j = idx & 127;
    int ad = (i > j) ? (i - j) : (j - i);
    int dist = (ad < 128 - ad) ? ad : (128 - ad);
    float z = (float)dist / 5.0f;
    float z2 = z * z, z3 = z2 * z, z4 = z2 * z2, z5 = z4 * z;
    float g;
    if (z < 1.0f)      g = 1.0f - (5.0f/3.0f)*z2 + (5.0f/8.0f)*z3 + 0.5f*z4 - 0.25f*z5;
    else if (z < 2.0f) g = 4.0f - 5.0f*z + (5.0f/3.0f)*z2 + (5.0f/8.0f)*z3 - 0.5f*z4
                           + (1.0f/12.0f)*z5 - 2.0f/(3.0f*z);
    else               g = 0.0f;
    taper[idx] = g;
    B1[swz_idx(i, j, 8)] = (_Float16)W1[j * 128 + i];   // B1[k=i][n=j] = W1[n][k]
  }
  if (idx < 8192) { int k = idx >> 6, n = idx & 63;  B2[swz_idx(k, n, 4)] = (_Float16)W2[n * 128 + k]; }
  if (idx < 2048) { int k = idx >> 5, n = idx & 31;  B3[swz_idx(k, n, 2)] = (_Float16)W3[n * 64 + k]; }
  if (idx < 4096) { int k = idx >> 7, n = idx & 127; B4[swz_idx(k, n, 8)] = (_Float16)W4[n * 32 + k]; }
}

// =====================  main: one workgroup per batch element, full recursion in LDS  ==========
__global__ __launch_bounds__(256)
void dssm_enkf_kernel(const float* __restrict__ obs,
                      const float* __restrict__ b1g, const float* __restrict__ b2g,
                      const float* __restrict__ b3g, const float* __restrict__ b4g,
                      const float* __restrict__ Wih, const float* __restrict__ Whh,
                      const float* __restrict__ bihg,
                      const float* __restrict__ Wm, const float* __restrict__ bmg,
                      const float* __restrict__ Wv, const float* __restrict__ bvg,
                      const float* __restrict__ eps0, const float* __restrict__ eps_seq,
                      const float* __restrict__ noise_y,
                      const float* __restrict__ taper,
                      const uint4* __restrict__ wsw,   // 61440 B of swizzled f16 weights
                      float* __restrict__ kl_arr, float* __restrict__ ll_arr,
                      float* __restrict__ out_means) {
  extern __shared__ char smem[];
  float*    X     = (float*)(smem + X_OFF);     // [112][132] ensemble (padded rows zero)
  _Float16* a16   = (_Float16*)(smem + A_OFF);  // MLP input staging
  float*    innov = (float*)(smem + A_OFF);     // later: [112][68]
  _Float16* h1    = (_Float16*)(smem + H1_OFF);
  float*    S     = (float*)(smem + H1_OFF);    // later: [64][65] S -> Cholesky L
  _Float16* h2    = (_Float16*)(smem + H2_OFF);
  _Float16* h3    = (_Float16*)(smem + H3_OFF);
  float*    Km    = (float*)(smem + H2_OFF);    // later: [64][132] HP -> Kalman gain
  float* xm   = (float*)(smem + SM_XM);
  float* qm   = (float*)(smem + SM_QM);
  float* qv   = (float*)(smem + SM_QV);
  float* yv   = (float*)(smem + SM_YV);
  float* yp   = (float*)(smem + SM_YP);
  float* hls  = (float*)(smem + SM_H);
  float* cls  = (float*)(smem + SM_C);
  float* xls  = (float*)(smem + SM_XL);
  float* gat  = (float*)(smem + SM_G);
  float* red  = (float*)(smem + SM_RED);
  float* zbuf = (float*)(smem + SM_Z);
  float* bb1  = (float*)(smem + SM_B1);
  float* bb2  = (float*)(smem + SM_B2);
  float* bb3  = (float*)(smem + SM_B3);
  float* bb4  = (float*)(smem + SM_B4);
  float* bih  = (float*)(smem + SM_BIH);
  _Float16* WB1 = (_Float16*)(smem + WL1_OFF);
  _Float16* WB2 = (_Float16*)(smem + WL2_OFF);
  _Float16* WB3 = (_Float16*)(smem + WL3_OFF);
  _Float16* WB4 = (_Float16*)(smem + WL4_OFF);
  float*    epsl = (float*)(smem + EPS_OFF);    // [100][128]

  const int tid  = threadIdx.x;
  const int b    = blockIdx.x;
  const int lane = tid & 31;
  const int wave = tid >> 5;

  // stage swizzled weights (61440 B) and biases once; reused all 50 steps
  {
    uint4* dst = (uint4*)(smem + WL1_OFF);
    for (int i = tid; i < 3840; i += 256) dst[i] = wsw[i];
  }
  if (tid < 128) bb1[tid] = b1g[tid];
  if (tid < 64)  bb2[tid] = b2g[tid];
  if (tid < 32)  bb3[tid] = b3g[tid];
  if (tid < 128) bb4[tid] = b4g[tid];
  bih[tid] = bihg[tid];
  if (tid < 64) { hls[tid] = 0.0f; cls[tid] = 0.0f; }
  __syncthreads();

  // ---------------- LSTM recognition network (sequential, tiny) ----------------
  for (int t = 0; t < 50; ++t) {
    if (tid < 64) xls[tid] = obs[(b * 50 + t) * 64 + tid];
    __syncthreads();
    {
      float acc = bih[tid];
      const float* wi = Wih + tid * 64;
      const float* wh = Whh + tid * 64;
#pragma unroll 4
      for (int k = 0; k < 64; ++k) acc += wi[k] * xls[k] + wh[k] * hls[k];
      gat[tid] = acc;
    }
    __syncthreads();
    if (tid < 64) {
      float ig = gat[tid], fg = gat[64 + tid], gg = gat[128 + tid], og = gat[192 + tid];
      float c = sigmoidf_(fg) * cls[tid] + sigmoidf_(ig) * tanhf(gg);
      cls[tid] = c;
      hls[tid] = sigmoidf_(og) * tanhf(c);
    }
    __syncthreads();
  }

  // ---------------- q(x0), KL, ensemble init ----------------
  if (tid < 128) {
    float am = bmg[tid], av = bvg[tid];
    const float* wm = Wm + tid * 64;
    const float* wv = Wv + tid * 64;
    for (int k = 0; k < 64; ++k) { am += wm[k] * hls[k]; av += wv[k] * hls[k]; }
    qm[tid] = am;
    float sp = (av > 20.0f) ? av : log1pf(__expf(av));
    qv[tid] = sp + 1e-4f;
  }
  __syncthreads();
  red[tid] = (tid < 128) ? (qv[tid] + qm[tid] * qm[tid] - 1.0f - logf(qv[tid])) : 0.0f;
  __syncthreads();
  for (int s = 128; s > 0; s >>= 1) { if (tid < s) red[tid] += red[tid + s]; __syncthreads(); }
  if (tid == 0) kl_arr[b] = 0.5f * red[0];

  for (int idx = tid; idx < 100 * 128; idx += 256) {
    int m = idx >> 7, d = idx & 127;
    X[m * XS + d] = qm[d] + sqrtf(qv[d]) * eps0[((size_t)m * 128 + b) * 128 + d];
  }
  for (int idx = tid; idx < 12 * XS; idx += 256) X[100 * XS + idx] = 0.0f; // keep pad rows 0
  __syncthreads();

  // ================================ EnKF time recursion ================================
  for (int t = 0; t < 50; ++t) {
    // ---- async prefetch of this step's eps tile (100 rows x 512 B) into LDS ----
    // Overlaps the HBM fetch (dominant traffic, 327 MB total) with MLP layers 1-3.
    {
      const float* gbase = eps_seq + (((size_t)t * 100) * 128 + b) * 128;
      for (int idx = tid; idx < 3200; idx += 256) {   // 3200 x b128 = 51200 B
        int m = idx >> 5, q = idx & 31;
        const float* g = gbase + (size_t)m * (128 * 128) + q * 4;
        unsigned lo = (unsigned)(EPS_OFF + (m * 128 + q * 4) * 4);
        asm volatile("global_load_async_to_lds_b128 %0, %1, off"
                     :: "v"(lo), "v"(g) : "memory");
      }
    }

    // ---- stage ensemble as f16 for the MLP ----
    for (int idx = tid; idx < NMC_PAD * 128; idx += 256) {
      int m = idx >> 7, k = idx & 127;
      a16[m * AS + k] = (_Float16)X[m * XS + k];
    }
    __syncthreads();

    // ---- layer1: h1 = relu(a16 @ W1^T + b1)  [112 x 128], K=128 ----
    for (int tile = wave; tile < 56; tile += 8) {
      int mb = (tile >> 3) * 16, nb = tile & 7;
      v8f acc = {};
#pragma unroll
      for (int kc = 0; kc < 4; ++kc) {
        v16h a = load_a16f(a16, AS, mb, kc * 32, lane);
        v16h bf = load_b16sw(WB1, 8, kc, nb, lane);
        acc = __builtin_amdgcn_wmma_f32_16x16x32_f16(false, a, false, bf, (short)0, acc, false, false);
      }
      int cc = lane & 15, hi = lane >> 4, nb16 = nb * 16;
#pragma unroll
      for (int v = 0; v < 8; ++v) {
        int row = mb + v + 8 * hi, col = nb16 + cc;
        h1[row * H1S + col] = (_Float16)fmaxf(acc[v] + bb1[col], 0.0f);
      }
    }
    __syncthreads();

    // ---- layer2: h2 = relu(h1 @ W2^T + b2)  [112 x 64], K=128 ----
    for (int tile = wave; tile < 28; tile += 8) {
      int mb = (tile >> 2) * 16, nb = tile & 3;
      v8f acc = {};
#pragma unroll
      for (int kc = 0; kc < 4; ++kc) {
        v16h a = load_a16f(h1, H1S, mb, kc * 32, lane);
        v16h bf = load_b16sw(WB2, 4, kc, nb, lane);
        acc = __builtin_amdgcn_wmma_f32_16x16x32_f16(false, a, false, bf, (short)0, acc, false, false);
      }
      int cc = lane & 15, hi = lane >> 4, nb16 = nb * 16;
#pragma unroll
      for (int v = 0; v < 8; ++v) {
        int row = mb + v + 8 * hi, col = nb16 + cc;
        h2[row * H2S + col] = (_Float16)fmaxf(acc[v] + bb2[col], 0.0f);
      }
    }
    __syncthreads();

    // ---- layer3: h3 = relu(h2 @ W3^T + b3)  [112 x 32], K=64 ----
    for (int tile = wave; tile < 14; tile += 8) {
      int mb = (tile >> 1) * 16, nb = tile & 1;
      v8f acc = {};
#pragma unroll
      for (int kc = 0; kc < 2; ++kc) {
        v16h a = load_a16f(h2, H2S, mb, kc * 32, lane);
        v16h bf = load_b16sw(WB3, 2, kc, nb, lane);
        acc = __builtin_amdgcn_wmma_f32_16x16x32_f16(false, a, false, bf, (short)0, acc, false, false);
      }
      int cc = lane & 15, hi = lane >> 4, nb16 = nb * 16;
#pragma unroll
      for (int v = 0; v < 8; ++v) {
        int row = mb + v + 8 * hi, col = nb16 + cc;
        h3[row * H3S + col] = (_Float16)fmaxf(acc[v] + bb3[col], 0.0f);
      }
    }
    // eps tile must be resident before the layer-4 epilogue consumes it
    asm volatile("s_wait_asynccnt 0x0" ::: "memory");
    __syncthreads();

    // ---- layer4 + process noise: X = h3 @ W4^T + b4 + 0.05*eps_t  [112 x 128], K=32 ----
    for (int tile = wave; tile < 56; tile += 8) {
      int mb = (tile >> 3) * 16, nb = tile & 7;
      v8f acc = {};
      {
        v16h a = load_a16f(h3, H3S, mb, 0, lane);
        v16h bf = load_b16sw(WB4, 8, 0, nb, lane);
        acc = __builtin_amdgcn_wmma_f32_16x16x32_f16(false, a, false, bf, (short)0, acc, false, false);
      }
      int cc = lane & 15, hi = lane >> 4, nb16 = nb * 16;
#pragma unroll
      for (int v = 0; v < 8; ++v) {
        int row = mb + v + 8 * hi, col = nb16 + cc;
        if (row < 100) {
          X[row * XS + col] = acc[v] + bb4[col] + 0.05f * epsl[row * 128 + col];
        } else {
          X[row * XS + col] = 0.0f;
        }
      }
    }
    __syncthreads();

    // ---- ensemble mean + center in place ----
    if (tid < 128) {
      float s = 0.0f;
      for (int m = 0; m < 100; ++m) s += X[m * XS + tid];
      xm[tid] = s * (1.0f / 100.0f);
    }
    if (tid < 64) {
      float y = obs[(b * 50 + t) * 64 + tid];
      yv[tid] = y;
      yp[tid] = y + 0.1f * noise_y[((size_t)t * 128 + b) * 64 + tid];
    }
    __syncthreads();
    for (int idx = tid; idx < 100 * 128; idx += 256) {
      int m = idx >> 7, k = idx & 127;
      X[m * XS + k] -= xm[k];
    }
    __syncthreads();

    // ---- localized covariance rows: HP = taper[:64,:] * (Xc^T Xc / 99); S = HP[:,:64] + R ----
    for (int tile = wave; tile < 32; tile += 8) {
      int ob = (tile >> 3) * 16, eb = (tile & 7) * 16;
      v8f acc = {};
      for (int kc = 0; kc < 25; ++kc) {           // K = 100 = 25 x 4
        v2f a = load_a32t(X, XS, ob, kc * 4, lane);
        v2f bf = load_b32f(X, XS, kc * 4, eb, lane);
        acc = __builtin_amdgcn_wmma_f32_16x16x4_f32(false, a, false, bf, (short)0, acc, false, false);
      }
      int cc = lane & 15, hi = lane >> 4;
#pragma unroll
      for (int v = 0; v < 8; ++v) {
        int o = ob + v + 8 * hi, e = eb + cc;
        float val = acc[v] * (1.0f / 99.0f) * taper[o * 128 + e];
        Km[o * KS + e] = val;
        if (e < 64) S[o * SS + e] = val + ((o == e) ? 0.01f : 0.0f);
      }
    }
    __syncthreads();

    // ---- Cholesky S = L L^T (in place, lower) ----
    for (int k = 0; k < 64; ++k) {
      if (tid == 0) S[k * SS + k] = sqrtf(S[k * SS + k]);
      __syncthreads();
      if (tid > k && tid < 64) S[tid * SS + k] /= S[k * SS + k];
      __syncthreads();
      for (int idx = tid; idx < 64 * 64; idx += 256) {
        int i = idx >> 6, j = idx & 63;
        if (i > k && j > k && j <= i) S[i * SS + j] -= S[i * SS + k] * S[j * SS + k];
      }
      __syncthreads();
    }

    // ---- K = S^-1 HP via two triangular solves, one RHS column per thread ----
    if (tid < 128) {
      int c = tid;
      for (int i = 0; i < 64; ++i) {
        float v = Km[i * KS + c];
        for (int j = 0; j < i; ++j) v -= S[i * SS + j] * Km[j * KS + c];
        Km[i * KS + c] = v / S[i * SS + i];
      }
      for (int i = 63; i >= 0; --i) {
        float v = Km[i * KS + c];
        for (int j = i + 1; j < 64; ++j) v -= S[j * SS + i] * Km[j * KS + c];
        Km[i * KS + c] = v / S[i * SS + i];
      }
    }
    __syncthreads();

    // ---- innovation (vs centered ensemble): innov[m][o] = yp[o] - xm[o] - Xc[m][o] ----
    for (int idx = tid; idx < NMC_PAD * 64; idx += 256) {
      int m = idx >> 6, o = idx & 63;
      innov[m * INS + o] = (m < 100) ? (yp[o] - xm[o] - X[m * XS + o]) : 0.0f;
    }
    __syncthreads();

    // ---- posterior ensemble: X = Xc + innov @ K + xm  [112 x 128], K=64 ----
    for (int tile = wave; tile < 56; tile += 8) {
      int mb = (tile >> 3) * 16, nb = (tile & 7) * 16;
      int cc = lane & 15, hi = lane >> 4;
      v8f acc;
#pragma unroll
      for (int v = 0; v < 8; ++v) acc[v] = X[(mb + v + 8 * hi) * XS + nb + cc];
#pragma unroll
      for (int kc = 0; kc < 16; ++kc) {           // K = 64 = 16 x 4
        v2f a = load_a32f(innov, INS, mb, kc * 4, lane);
        v2f bf = load_b32f(Km, KS, kc * 4, nb, lane);
        acc = __builtin_amdgcn_wmma_f32_16x16x4_f32(false, a, false, bf, (short)0, acc, false, false);
      }
#pragma unroll
      for (int v = 0; v < 8; ++v) {
        int row = mb + v + 8 * hi, col = nb + cc;
        X[row * XS + col] = (row < 100) ? (acc[v] + xm[col]) : 0.0f;
      }
    }
    __syncthreads();

    // ---- posterior mean output + EnKF marginal log-likelihood ----
    if (tid < 128) {
      float acc = xm[tid];
      for (int o = 0; o < 64; ++o) acc += (yv[o] - xm[o]) * Km[o * KS + tid];
      out_means[((size_t)t * 128 + b) * 128 + tid] = acc;
    }
    if (tid == 0) {
      float logdet = 0.0f, s2 = 0.0f;
      for (int i = 0; i < 64; ++i) {
        float v = yv[i] - xm[i];
        for (int j = 0; j < i; ++j) v -= S[i * SS + j] * zbuf[j];
        float zi = v / S[i * SS + i];
        zbuf[i] = zi;
        logdet += logf(S[i * SS + i]);
        s2 += zi * zi;
      }
      ll_arr[t * 128 + b] = -0.5f * s2 - logdet - 32.0f * 1.8378770664093453f; // 0.5*DY*log(2pi)
    }
    __syncthreads();
  }
}

// =====================  finalize: elbo = (sum ll - sum kl) / B  =====================
__global__ void dssm_final_kernel(const float* __restrict__ kl_arr,
                                  const float* __restrict__ ll_arr, float* __restrict__ out) {
  __shared__ float red[256];
  int tid = threadIdx.x;
  float s = 0.0f;
  for (int i = tid; i < 50 * 128; i += 256) s += ll_arr[i];
  if (tid < 128) s -= kl_arr[tid];
  red[tid] = s;
  __syncthreads();
  for (int st = 128; st > 0; st >>= 1) { if (tid < st) red[tid] += red[tid + st]; __syncthreads(); }
  if (tid == 0) out[0] = red[0] * (1.0f / 128.0f);
}

extern "C" void kernel_launch(void* const* d_in, const int* in_sizes, int n_in,
                              void* d_out, int out_size, void* d_ws, size_t ws_size,
                              hipStream_t stream) {
  (void)in_sizes; (void)n_in; (void)out_size; (void)ws_size;
  const float* obs   = (const float*)d_in[0];
  const float* W1    = (const float*)d_in[1];
  const float* b1    = (const float*)d_in[2];
  const float* W2    = (const float*)d_in[3];
  const float* b2    = (const float*)d_in[4];
  const float* W3    = (const float*)d_in[5];
  const float* b3    = (const float*)d_in[6];
  const float* W4    = (const float*)d_in[7];
  const float* b4    = (const float*)d_in[8];
  const float* Wih   = (const float*)d_in[9];
  const float* Whh   = (const float*)d_in[10];
  const float* bih   = (const float*)d_in[11];
  const float* Wm    = (const float*)d_in[12];
  const float* bm    = (const float*)d_in[13];
  const float* Wv    = (const float*)d_in[14];
  const float* bv    = (const float*)d_in[15];
  const float* eps0  = (const float*)d_in[16];
  const float* epsq  = (const float*)d_in[17];
  const float* ny    = (const float*)d_in[18];

  char* ws = (char*)d_ws;
  float*    taper = (float*)(ws + WS_TAPER);
  _Float16* B1    = (_Float16*)(ws + WS_B1);
  _Float16* B2    = (_Float16*)(ws + WS_B2);
  _Float16* B3    = (_Float16*)(ws + WS_B3);
  _Float16* B4    = (_Float16*)(ws + WS_B4);
  float*    klar  = (float*)(ws + WS_KL);
  float*    llar  = (float*)(ws + WS_LL);
  float*    out   = (float*)d_out;

  (void)hipFuncSetAttribute((const void*)dssm_enkf_kernel,
                            hipFuncAttributeMaxDynamicSharedMemorySize, SMEM_TOTAL);

  dssm_prep_kernel<<<64, 256, 0, stream>>>(W1, W2, W3, W4, taper, B1, B2, B3, B4);
  dssm_enkf_kernel<<<128, 256, SMEM_TOTAL, stream>>>(
      obs, b1, b2, b3, b4, Wih, Whh, bih, Wm, bm, Wv, bv,
      eps0, epsq, ny, taper, (const uint4*)(ws + WS_B1), klar, llar, out + 1);
  dssm_final_kernel<<<1, 256, 0, stream>>>(klar, llar, out);
}
